// FFNwMoE_40913858461728
// MI455X (gfx1250) — compile-verified, matrix operand order checked by reading, not run on hardware
//
#include <hip/hip_runtime.h>
#include <hip/hip_bf16.h>

// ---------------------------------------------------------------------------
// MoE SwiGLU FFN for MI455X (gfx1250, wave32, WMMA bf16).
//   out[n,:] = sum_{i in top2(n)} gate_i * ( silu(x W1_e^T) * (x W3_e^T) ) W2_e^T
// Strategy: bf16 weight cache in ws (100MB -> L2-resident), grouped GEMM over
// per-expert gathered 32-token tiles using v_wmma_f32_16x16x32_bf16.
// AI = 32 FLOP/byte of weight traffic (vs 16 at 16-token tiles).
// ---------------------------------------------------------------------------

typedef __attribute__((ext_vector_type(4)))  __bf16 v4bf;
typedef __attribute__((ext_vector_type(8)))  __bf16 v8bf;
typedef __attribute__((ext_vector_type(16))) __bf16 v16bf;
typedef __attribute__((ext_vector_type(8)))  float  v8f;

#define NTOK 4096          // B*S tokens
#define DDIM 1024
#define HDIM 2048
#define NEXP 8
#define TILE_M 32          // tokens per workgroup tile
#define TILES_PER_E 128    // NTOK / TILE_M
#define HCHUNK 512         // h columns per chunk (16 waves * 32)
#define XS_STR 1032        // 1024 + 8 pad (bank spread, keeps 16B row align)
#define HS_STR 520         // 512 + 8 pad

static __device__ __forceinline__ v16bf ldab(const __bf16* p) {
  // A/B fragment: 8 bf16 at p (K = k0+koff .. +7) and 8 bf16 at p+16
  // (K = k0+16+koff .. +23) per the 16-bit 16x32 operand layout.
  v8bf lo = *(const v8bf*)p;
  v8bf hi = *(const v8bf*)(p + 16);
  return __builtin_shufflevector(lo, hi, 0,1,2,3,4,5,6,7,8,9,10,11,12,13,14,15);
}

static __device__ __forceinline__ v8f wmma_bf16(v16bf a, v16bf b, v8f c) {
  return __builtin_amdgcn_wmma_f32_16x16x32_bf16(false, a, false, b,
                                                 (short)0, c, false, false);
}

// --------------------------- small helper kernels ---------------------------

__global__ void init_ctrl_kernel(int* ctrl) {
  if (threadIdx.x < 32) ctrl[threadIdx.x] = 0;   // counts, top1cnt, probsum, offsets
}

__global__ __launch_bounds__(256) void cvt_bf16_kernel(const float4* __restrict__ src,
                                                       v4bf* __restrict__ dst, int n4) {
  int i = blockIdx.x * 256 + threadIdx.x;
  if (i >= n4) return;
  float4 v = src[i];
  v4bf o;
  o[0] = (__bf16)v.x; o[1] = (__bf16)v.y; o[2] = (__bf16)v.z; o[3] = (__bf16)v.w;
  dst[i] = o;
}

// Router: one wave32 per token. Coalesced reads of x, Wr stays in cache.
__global__ __launch_bounds__(256) void router_kernel(
    const float* __restrict__ x, const float* __restrict__ Wr,
    int* __restrict__ counts, int* __restrict__ top1cnt, float* __restrict__ probsum,
    int* __restrict__ lists, float* __restrict__ gates_tok, int* __restrict__ tokmap) {
  const int lane = threadIdx.x & 31;
  const int n = blockIdx.x * 8 + (threadIdx.x >> 5);
  float acc[NEXP];
#pragma unroll
  for (int e = 0; e < NEXP; ++e) acc[e] = 0.f;
  const float* xr = x + (size_t)n * DDIM;
  for (int i = lane; i < DDIM; i += 32) {
    float xv = xr[i];
#pragma unroll
    for (int e = 0; e < NEXP; ++e) acc[e] += xv * Wr[e * DDIM + i];
  }
#pragma unroll
  for (int e = 0; e < NEXP; ++e) {
#pragma unroll
    for (int off = 16; off > 0; off >>= 1) acc[e] += __shfl_xor(acc[e], off, 32);
  }
  if (lane == 0) {
    float mx = acc[0];
#pragma unroll
    for (int e = 1; e < NEXP; ++e) mx = fmaxf(mx, acc[e]);
    float p[NEXP], sum = 0.f;
#pragma unroll
    for (int e = 0; e < NEXP; ++e) { p[e] = __expf(acc[e] - mx); sum += p[e]; }
    float inv = 1.f / sum;
#pragma unroll
    for (int e = 0; e < NEXP; ++e) p[e] *= inv;
    int i1 = 0;
#pragma unroll
    for (int e = 1; e < NEXP; ++e) if (acc[e] > acc[i1]) i1 = e;   // first-index ties
    int i2 = (i1 == 0) ? 1 : 0;
#pragma unroll
    for (int e = 0; e < NEXP; ++e) if (e != i1 && acc[e] > acc[i2]) i2 = e;
    int pos1 = atomicAdd(&counts[i1], 1);
    int pos2 = atomicAdd(&counts[i2], 1);
    lists[i1 * NTOK + pos1] = n;
    lists[i2 * NTOK + pos2] = n;
    tokmap[2 * n]     = (i1 << 16) | pos1;
    tokmap[2 * n + 1] = (i2 << 16) | pos2;
    gates_tok[2 * n]     = p[i1];
    gates_tok[2 * n + 1] = p[i2];
    atomicAdd(&top1cnt[i1], 1);
#pragma unroll
    for (int e = 0; e < NEXP; ++e) atomicAdd(&probsum[e], p[e]);
  }
}

__global__ void prefix_aux_kernel(const int* counts, const int* top1cnt,
                                  const float* probsum, int* offsets, float* out_aux) {
  if (threadIdx.x == 0 && blockIdx.x == 0) {
    int o = 0;
    float aux = 0.f;
    for (int e = 0; e < NEXP; ++e) {
      offsets[e] = o;
      o += counts[e];
      aux += ((float)top1cnt[e] / (float)NTOK) * (probsum[e] / (float)NTOK);
    }
    *out_aux = 0.01f * aux * (float)NEXP;
  }
}

// ------------------------------- FFN (WMMA) --------------------------------
// Block = 16 waves (512 threads). Tile = 32 gathered tokens x full D output.
// Per 512-wide H chunk: each wave builds its own 32 h-columns (silu*mul fused
// in the f32 C layout, 2 M-subtiles), shares them via LDS, then all waves
// advance the y = h @ W2^T accumulators (32x64 per wave, 64 VGPRs), with each
// W2 fragment reused across both A M-subtiles.
__global__ __launch_bounds__(512) void ffn_kernel(
    const float* __restrict__ x,
    const __bf16* __restrict__ W1b, const __bf16* __restrict__ W2b,
    const __bf16* __restrict__ W3b,
    const int* __restrict__ counts, const int* __restrict__ offsets,
    const int* __restrict__ lists, float* __restrict__ ybuf) {
  const int e    = blockIdx.x >> 7;          // TILES_PER_E == 128
  const int tile = blockIdx.x & 127;
  const int cnt  = counts[e];
  if (tile * TILE_M >= cnt) return;

  __shared__ __align__(16) __bf16 xs[TILE_M * XS_STR];
  __shared__ __align__(16) __bf16 hs[TILE_M * HS_STR];
  __shared__ int tok_s[TILE_M];

  const int tid  = threadIdx.x;
  const int lane = tid & 31;
  const int wave = tid >> 5;

  if (tid < TILE_M) {
    int r = tile * TILE_M + tid;
    tok_s[tid] = (r < cnt) ? lists[e * NTOK + r] : -1;
  }
  __syncthreads();

  // Stage gathered x rows into LDS as bf16 (coalesced float4 reads, 2 rows/iter).
  {
    const int col   = (tid & 255) * 4;
    const int rhalf = tid >> 8;              // 0 or 1
    for (int it = 0; it < TILE_M / 2; ++it) {
      int row = it * 2 + rhalf;
      int tok = tok_s[row];
      v4bf o;
      if (tok >= 0) {
        float4 v = *(const float4*)(x + (size_t)tok * DDIM + col);
        o[0] = (__bf16)v.x; o[1] = (__bf16)v.y; o[2] = (__bf16)v.z; o[3] = (__bf16)v.w;
      } else {
        o[0] = o[1] = o[2] = o[3] = (__bf16)0.f;
      }
      *(v4bf*)(xs + row * XS_STR + col) = o;
    }
  }
  __syncthreads();

  v8f acc[2][4] = {};                     // y tile: 32 rows x 64 cols (this wave)
  const int m     = lane & 15;            // A row / B column within subtile
  const int koff  = (lane >> 4) * 8;      // K half select per operand layout
  const int dcol0 = wave * 64;

  const __bf16* pa0 = xs + m * XS_STR + koff;
  const __bf16* pa1 = xs + (16 + m) * XS_STR + koff;

  for (int hc = 0; hc < HDIM; hc += HCHUNK) {
    // ---- phase 1: h[:, hc + wave*32 .. +32) = silu(x W1^T) * (x W3^T) ----
    {
      v8f c1[2][2] = {}, c3[2][2] = {};   // [m-subtile][n-subtile]
      const size_t wrow = (size_t)(e * HDIM + hc + wave * 32 + m) * DDIM + koff;
      const __bf16* p1a = W1b + wrow;
      const __bf16* p1b = p1a + 16 * DDIM;
      const __bf16* p3a = W3b + wrow;
      const __bf16* p3b = p3a + 16 * DDIM;
#pragma unroll 2
      for (int k0 = 0; k0 < DDIM; k0 += 32) {
        v16bf a0 = ldab(pa0 + k0);
        v16bf a1 = ldab(pa1 + k0);
        v16bf b;
        b = ldab(p1a + k0);
        c1[0][0] = wmma_bf16(a0, b, c1[0][0]);
        c1[1][0] = wmma_bf16(a1, b, c1[1][0]);
        b = ldab(p1b + k0);
        c1[0][1] = wmma_bf16(a0, b, c1[0][1]);
        c1[1][1] = wmma_bf16(a1, b, c1[1][1]);
        b = ldab(p3a + k0);
        c3[0][0] = wmma_bf16(a0, b, c3[0][0]);
        c3[1][0] = wmma_bf16(a1, b, c3[1][0]);
        b = ldab(p3b + k0);
        c3[0][1] = wmma_bf16(a0, b, c3[0][1]);
        c3[1][1] = wmma_bf16(a1, b, c3[1][1]);
      }
      // C layout: VGPR r -> row r + 8*(lane>=16), col = lane&15.
      const int hrow0 = (lane >> 4) * 8;
      const int hcol  = wave * 32 + (lane & 15);
#pragma unroll
      for (int mi = 0; mi < 2; ++mi) {
#pragma unroll
        for (int r = 0; r < 8; ++r) {
          const int row = mi * 16 + hrow0 + r;
          float g  = c1[mi][0][r];
          hs[row * HS_STR + hcol]      = (__bf16)((g  / (1.f + __expf(-g)))  * c3[mi][0][r]);
          float g2 = c1[mi][1][r];
          hs[row * HS_STR + hcol + 16] = (__bf16)((g2 / (1.f + __expf(-g2))) * c3[mi][1][r]);
        }
      }
    }
    __syncthreads();
    // ---- phase 2: acc += h[32,512] @ W2[dcol0..+64, hc..+512]^T ----
    {
      const __bf16* ph0 = hs + m * HS_STR + koff;
      const __bf16* ph1 = hs + (16 + m) * HS_STR + koff;
      const __bf16* p2  = W2b + (size_t)(e * DDIM + dcol0 + m) * HDIM + hc + koff;
      for (int k0 = 0; k0 < HCHUNK; k0 += 32) {
        v16bf a0 = ldab(ph0 + k0);
        v16bf a1 = ldab(ph1 + k0);
#pragma unroll
        for (int s = 0; s < 4; ++s) {
          v16bf b = ldab(p2 + (size_t)s * 16 * HDIM + k0);
          acc[0][s] = wmma_bf16(a0, b, acc[0][s]);
          acc[1][s] = wmma_bf16(a1, b, acc[1][s]);
        }
      }
    }
    __syncthreads();
  }

  // Deterministic writeback: each (expert,pos) slot owned by exactly one block.
  const int base  = offsets[e] + tile * TILE_M;
  const int crow0 = (lane >> 4) * 8;
  const int ccol  = lane & 15;
#pragma unroll
  for (int mi = 0; mi < 2; ++mi) {
#pragma unroll
    for (int s = 0; s < 4; ++s) {
      const int dcol = dcol0 + s * 16 + ccol;
#pragma unroll
      for (int r = 0; r < 8; ++r) {
        const int row = mi * 16 + crow0 + r;
        if (tile * TILE_M + row < cnt)
          ybuf[(size_t)(base + row) * DDIM + dcol] = acc[mi][s][r];
      }
    }
  }
}

// Per-token gather of its two expert rows (deterministic combine).
__global__ __launch_bounds__(256) void combine_kernel(
    const float* __restrict__ ybuf, const int* __restrict__ offsets,
    const int* __restrict__ tokmap, const float* __restrict__ gates_tok,
    float* __restrict__ out) {
  const int idx = blockIdx.x * 256 + threadIdx.x;  // over N * (D/4)
  const int n   = idx >> 8;                        // D/4 == 256
  const int c   = (idx & 255) * 4;
  const int m0 = tokmap[2 * n], m1 = tokmap[2 * n + 1];
  const float g0 = gates_tok[2 * n], g1 = gates_tok[2 * n + 1];
  const int s0 = offsets[m0 >> 16] + (m0 & 0xFFFF);
  const int s1 = offsets[m1 >> 16] + (m1 & 0xFFFF);
  const float4 y0 = *(const float4*)(ybuf + (size_t)s0 * DDIM + c);
  const float4 y1 = *(const float4*)(ybuf + (size_t)s1 * DDIM + c);
  float4 o;
  o.x = g0 * y0.x + g1 * y1.x;
  o.y = g0 * y0.y + g1 * y1.y;
  o.z = g0 * y0.z + g1 * y1.z;
  o.w = g0 * y0.w + g1 * y1.w;
  *(float4*)(out + (size_t)n * DDIM + c) = o;
}

// --------------------------------- launch ----------------------------------

extern "C" void kernel_launch(void* const* d_in, const int* in_sizes, int n_in,
                              void* d_out, int out_size, void* d_ws, size_t ws_size,
                              hipStream_t stream) {
  (void)in_sizes; (void)n_in; (void)out_size; (void)ws_size;
  const float* x  = (const float*)d_in[0];
  const float* Wr = (const float*)d_in[1];
  const float* W1 = (const float*)d_in[2];
  const float* W2 = (const float*)d_in[3];
  const float* W3 = (const float*)d_in[4];
  float* out = (float*)d_out;

  const size_t SZW = (size_t)NEXP * HDIM * DDIM;        // elements per weight tensor
  char* ws = (char*)d_ws;
  __bf16* W1b  = (__bf16*)(ws);
  __bf16* W2b  = (__bf16*)(ws + SZW * 2);
  __bf16* W3b  = (__bf16*)(ws + SZW * 4);
  float*  ybuf = (float*)(ws + SZW * 6);                // 2*N x D f32
  char* ctrl = ws + SZW * 6 + (size_t)2 * NTOK * DDIM * 4;
  int*   counts    = (int*)ctrl;                        // [8]
  int*   top1cnt   = counts + 8;                        // [8]
  float* probsum   = (float*)(counts + 16);             // [8]
  int*   offsets   = counts + 24;                       // [8]
  int*   lists     = (int*)(ctrl + 256);                // [E*N]
  float* gates_tok = (float*)(ctrl + 256 + (size_t)NEXP * NTOK * 4);  // [2*N]
  int*   tokmap    = (int*)(ctrl + 256 + (size_t)NEXP * NTOK * 4 + (size_t)2 * NTOK * 4);

  init_ctrl_kernel<<<1, 32, 0, stream>>>(counts);

  const int n4 = (int)(SZW / 4);
  const int cvtBlocks = (n4 + 255) / 256;
  cvt_bf16_kernel<<<cvtBlocks, 256, 0, stream>>>((const float4*)W1, (v4bf*)W1b, n4);
  cvt_bf16_kernel<<<cvtBlocks, 256, 0, stream>>>((const float4*)W2, (v4bf*)W2b, n4);
  cvt_bf16_kernel<<<cvtBlocks, 256, 0, stream>>>((const float4*)W3, (v4bf*)W3b, n4);

  router_kernel<<<NTOK / 8, 256, 0, stream>>>(x, Wr, counts, top1cnt, probsum,
                                              lists, gates_tok, tokmap);
  prefix_aux_kernel<<<1, 1, 0, stream>>>(counts, top1cnt, probsum, offsets,
                                         out + (size_t)NTOK * DDIM);
  ffn_kernel<<<NEXP * TILES_PER_E, 512, 0, stream>>>(x, W1b, W2b, W3b, counts,
                                                     offsets, lists, ybuf);
  combine_kernel<<<NTOK, 256, 0, stream>>>(ybuf, offsets, tokmap, gates_tok, out);
}
